// MultiHeadAttention_41747082117504
// MI455X (gfx1250) — matrix operand order repaired and checked by reading, not compiled, not run on hardware
//
#include <hip/hip_runtime.h>
#include <hip/hip_bf16.h>

// ---------------------------------------------------------------------------
// MHA forward for MI455X (gfx1250, wave32, WMMA).
//   D_MODEL=1024, H=16, DEPTH=64, B=2, S=2048  -> M = B*S = 4096
// Outputs (concatenated in d_out): output [2,2048,1024] f32, weights [2,16,2048,2048] f32
// ---------------------------------------------------------------------------

typedef __attribute__((ext_vector_type(16))) __bf16 v16bf;
typedef __attribute__((ext_vector_type(8)))  float  v8f;

#define D_MODEL 1024
#define N_HEADS 16
#define DEPTH   64
#define SEQ     2048
#define MTOT    4096            // B*S

__device__ __forceinline__ v8f wmma_bf16(v16bf a, v16bf b, v8f c) {
  // D = A(16x32 bf16) * B(32x16 bf16) + C(16x16 f32)
  return __builtin_amdgcn_wmma_f32_16x16x32_bf16(
      /*neg_a=*/false, a, /*neg_b=*/false, b,
      /*c_mod=*/(short)0, c, /*reuse_a=*/false, /*reuse_b=*/false);
}

// A fragment (16x32, bf16): lane<16 -> M=lane, K base 0; lane>=16 -> M=lane-16, K base 16.
// elements 0..15 = K{klo+0..klo+15} (16 consecutive K values per lane).
__device__ __forceinline__ v16bf a_frag_f32(const float* __restrict__ row, int kb, int lane) {
  const int klo = (lane < 16) ? 0 : 16;
  v16bf r;
#pragma unroll
  for (int e = 0; e < 8; ++e) r[e]     = (__bf16)row[kb + klo + e];
#pragma unroll
  for (int e = 0; e < 8; ++e) r[8 + e] = (__bf16)row[kb + klo + 8 + e];
  return r;
}

__device__ __forceinline__ v16bf a_frag_bf16(const __bf16* __restrict__ row, int kb, int lane) {
  const int klo = (lane < 16) ? 0 : 16;
  v16bf r;
#pragma unroll
  for (int e = 0; e < 8; ++e) r[e]     = row[kb + klo + e];
#pragma unroll
  for (int e = 0; e < 8; ++e) r[8 + e] = row[kb + klo + 8 + e];
  return r;
}

// B fragment (32x16, bf16): lane column n = lane&15; lanes 0-15 hold K 0..15,
// lanes 16-31 hold K 16..31 -> 16 consecutive K values per lane.
__device__ __forceinline__ v16bf b_frag_f32(const float* __restrict__ row, int kb, int lane) {
  const int koff = kb + ((lane >> 4) << 4);
  v16bf r;
#pragma unroll
  for (int e = 0; e < 16; ++e) r[e] = (__bf16)row[koff + e];
  return r;
}

__device__ __forceinline__ v16bf b_frag_bf16(const __bf16* __restrict__ row, int kb, int lane) {
  const int koff = kb + ((lane >> 4) << 4);
  v16bf r;
#pragma unroll
  for (int e = 0; e < 16; ++e) r[e] = row[koff + e];
  return r;
}

// ---------------------------------------------------------------------------
// Kernel 1: projection GEMM  C[m,n] = sum_k X[m,k]*W[n,k] + bias[n]
// store modes: 0/1 -> bf16 [B,H,S,DEPTH]   (Q, K)
//              2   -> bf16 [B,H,DEPTH,S]   (V transposed, for P@V B-fragments)
// One 16x16 tile per wave, 8 waves/block, K-loop of 32.
// ---------------------------------------------------------------------------
__global__ __launch_bounds__(256) void mha_proj_kernel(
    const float* __restrict__ X, const float* __restrict__ W,
    const float* __restrict__ bias, void* dst_raw, int mode)
{
  __bf16* dst = (__bf16*)dst_raw;
  const int lane = threadIdx.x & 31;
  const int wave = threadIdx.x >> 5;
  const int tile = blockIdx.x * 8 + wave;          // 16384 tiles total
  const int m0 = (tile >> 6) << 4;                 // N/16 = 64
  const int n0 = (tile & 63) << 4;

  const float* arow = X + (size_t)(m0 + (lane & 15)) * D_MODEL;
  const float* brow = W + (size_t)(n0 + (lane & 15)) * D_MODEL;

  v8f acc = {};
  for (int kb = 0; kb < D_MODEL; kb += 32) {
    v16bf a = a_frag_f32(arow, kb, lane);
    v16bf b = b_frag_f32(brow, kb, lane);
    acc = wmma_bf16(a, b, acc);
  }

  const int mbase = m0 + ((lane < 16) ? 0 : 8);
  const int n = n0 + (lane & 15);
  const float bv = bias[n];
  const int h = n >> 6, d = n & 63;
#pragma unroll
  for (int r = 0; r < 8; ++r) {
    const int m = mbase + r;
    const int bb = m >> 11;                        // batch
    const int s  = m & (SEQ - 1);
    const float val = acc[r] + bv;
    size_t idx;
    if (mode == 2) idx = ((size_t)(bb * N_HEADS + h) * DEPTH + d) * SEQ + s;
    else           idx = ((size_t)(bb * N_HEADS + h) * SEQ + s) * DEPTH + d;
    dst[idx] = (__bf16)val;
  }
}

// ---------------------------------------------------------------------------
// Kernel 2: attention. One block per (b,h, 16-query-row stripe).
// 256 threads = 8 waves. LDS holds the full 16x2048 f32 score stripe (128 KB;
// CDNA5 has 320 KB LDS/WGP) so scores are computed, softmaxed, written to
// global exactly once, and consumed by P@V without leaving the WGP.
// ---------------------------------------------------------------------------
__global__ __launch_bounds__(256) void mha_attn_kernel(
    const void* Qraw, const void* Kraw, const void* VTraw,
    float* __restrict__ weights_out, void* attn_raw)
{
  const __bf16* Qb  = (const __bf16*)Qraw;
  const __bf16* Kb  = (const __bf16*)Kraw;
  const __bf16* VTb = (const __bf16*)VTraw;
  __bf16* attn_out  = (__bf16*)attn_raw;

  extern __shared__ char smem_raw[];
  float* sS      = (float*)smem_raw;              // [16][2048] scores / exp
  float* sAttn   = sS + 16 * SEQ;                 // [16][64] f32 reduction
  float* sRed    = sAttn + 16 * DEPTH;            // [16 seg][16 row] partials
  float* sRowMax = sRed + 256;                    // [16]
  float* sRowSum = sRowMax + 16;                  // [16]

  const int tid  = threadIdx.x;
  const int lane = tid & 31;
  const int wave = tid >> 5;
  const int bh   = blockIdx.x >> 7;               // 0..31 = b*16+h
  const int m0   = (blockIdx.x & 127) << 4;       // query row base

  const __bf16* Qh  = Qb  + (size_t)bh * SEQ * DEPTH;
  const __bf16* Kh  = Kb  + (size_t)bh * SEQ * DEPTH;
  const __bf16* VTh = VTb + (size_t)bh * DEPTH * SEQ;

  // ---- Phase 1: scores stripe  S = scale * Q * K^T  (wave owns 256 key cols)
  const __bf16* qrow = Qh + (size_t)(m0 + (lane & 15)) * DEPTH;
  const v16bf aq0 = a_frag_bf16(qrow, 0, lane);
  const v16bf aq1 = a_frag_bf16(qrow, 32, lane);
  const float scale = 0.125f;                     // 1/sqrt(64)
  const int mhalf = (lane < 16) ? 0 : 8;
  for (int ct = 0; ct < 16; ++ct) {
    const int n0 = wave * 256 + ct * 16;
    const __bf16* krow = Kh + (size_t)(n0 + (lane & 15)) * DEPTH;
    v16bf b0 = b_frag_bf16(krow, 0, lane);
    v16bf b1 = b_frag_bf16(krow, 32, lane);
    v8f acc = {};
    acc = wmma_bf16(aq0, b0, acc);
    acc = wmma_bf16(aq1, b1, acc);
    const int col = n0 + (lane & 15);
#pragma unroll
    for (int r = 0; r < 8; ++r)
      sS[(mhalf + r) * SEQ + col] = acc[r] * scale;
  }
  __syncthreads();

  // ---- Phase 2: softmax statistics (16 rows, 16 segments of 128 cols each)
  const int row = tid & 15;
  const int seg = tid >> 4;
  float* srow = sS + row * SEQ + seg * 128;
  float mx = -3.402823e38f;
  for (int i = 0; i < 128; ++i) mx = fmaxf(mx, srow[i]);
  sRed[seg * 16 + row] = mx;
  __syncthreads();
  if (tid < 16) {
    float m = sRed[tid];
    for (int s2 = 1; s2 < 16; ++s2) m = fmaxf(m, sRed[s2 * 16 + tid]);
    sRowMax[tid] = m;
  }
  __syncthreads();
  const float rm = sRowMax[row];
  float sum = 0.f;
  for (int i = 0; i < 128; ++i) {
    float e = __expf(srow[i] - rm);
    srow[i] = e;                                  // keep exp() in LDS
    sum += e;
  }
  sRed[seg * 16 + row] = sum;
  __syncthreads();
  if (tid < 16) {
    float s = 0.f;
    for (int s2 = 0; s2 < 16; ++s2) s += sRed[s2 * 16 + tid];
    sRowSum[tid] = s;
  }
  for (int i = tid; i < 16 * DEPTH; i += 256) sAttn[i] = 0.f;
  __syncthreads();

  // ---- Phase 3: single coalesced write of the normalized weights (537 MB total)
  float* wbase = weights_out + ((size_t)bh * SEQ + m0) * SEQ;
  for (int m = 0; m < 16; ++m) {
    const float inv = 1.0f / sRowSum[m];
    const float4* src4 = (const float4*)(sS + m * SEQ);
    float4* dst4 = (float4*)(wbase + (size_t)m * SEQ);
    for (int i = tid; i < SEQ / 4; i += 256) {
      float4 v = src4[i];
      v.x *= inv; v.y *= inv; v.z *= inv; v.w *= inv;
      dst4[i] = v;
    }
  }

  // ---- Phase 4: attn = P @ V, straight out of LDS. Wave owns a 256-wide k slice.
  {
    const int mA   = lane & 15;
    const int klo  = (lane < 16) ? 0 : 16;
    const int khlf = (lane >> 4) << 4;
    const float invA = 1.0f / sRowSum[mA];
    v8f acc[4] = {};
    for (int it = 0; it < 8; ++it) {
      const int kw = wave * 256 + it * 32;
      v16bf a;
#pragma unroll
      for (int e = 0; e < 8; ++e) a[e]     = (__bf16)(sS[mA * SEQ + kw + klo + e] * invA);
#pragma unroll
      for (int e = 0; e < 8; ++e) a[8 + e] = (__bf16)(sS[mA * SEQ + kw + klo + 8 + e] * invA);
#pragma unroll
      for (int nt = 0; nt < 4; ++nt) {
        const __bf16* vrow = VTh + (size_t)(nt * 16 + (lane & 15)) * SEQ + kw + khlf;
        v16bf b;
#pragma unroll
        for (int e = 0; e < 16; ++e) b[e] = vrow[e];
        acc[nt] = wmma_bf16(a, b, acc[nt]);
      }
    }
#pragma unroll
    for (int nt = 0; nt < 4; ++nt)
#pragma unroll
      for (int r = 0; r < 8; ++r)
        atomicAdd(&sAttn[(mhalf + r) * DEPTH + nt * 16 + (lane & 15)], acc[nt][r]);
  }
  __syncthreads();

  // ---- Phase 5: store context bf16 as [B, S, H*DEPTH] for the output GEMM
  const int bb = bh >> 4, hh = bh & 15;
  for (int i = tid; i < 16 * DEPTH; i += 256) {
    const int m = i >> 6, d = i & 63;
    attn_out[((size_t)bb * SEQ + (m0 + m)) * D_MODEL + hh * DEPTH + d] = (__bf16)sAttn[i];
  }
}

// ---------------------------------------------------------------------------
// Kernel 3: output projection  out[m,n] = sum_k A[m,k]*Wo[n,k] + bo[n], f32 out
// ---------------------------------------------------------------------------
__global__ __launch_bounds__(256) void mha_out_kernel(
    const void* Araw, const float* __restrict__ W,
    const float* __restrict__ bias, float* __restrict__ out)
{
  const __bf16* A = (const __bf16*)Araw;
  const int lane = threadIdx.x & 31;
  const int wave = threadIdx.x >> 5;
  const int tile = blockIdx.x * 8 + wave;
  const int m0 = (tile >> 6) << 4;
  const int n0 = (tile & 63) << 4;

  const __bf16* arow = A + (size_t)(m0 + (lane & 15)) * D_MODEL;
  const float*  brow = W + (size_t)(n0 + (lane & 15)) * D_MODEL;

  v8f acc = {};
  for (int kb = 0; kb < D_MODEL; kb += 32) {
    v16bf a = a_frag_bf16(arow, kb, lane);
    v16bf b = b_frag_f32(brow, kb, lane);
    acc = wmma_bf16(a, b, acc);
  }

  const int mbase = m0 + ((lane < 16) ? 0 : 8);
  const int n = n0 + (lane & 15);
  const float bv = bias[n];
#pragma unroll
  for (int r = 0; r < 8; ++r)
    out[(size_t)(mbase + r) * D_MODEL + n] = acc[r] + bv;
}

// ---------------------------------------------------------------------------
extern "C" void kernel_launch(void* const* d_in, const int* in_sizes, int n_in,
                              void* d_out, int out_size, void* d_ws, size_t ws_size,
                              hipStream_t stream) {
  (void)in_sizes; (void)n_in; (void)out_size; (void)ws_size;
  const float* q  = (const float*)d_in[0];
  const float* k  = (const float*)d_in[1];
  const float* v  = (const float*)d_in[2];
  const float* Wq = (const float*)d_in[3];
  const float* bq = (const float*)d_in[4];
  const float* Wk = (const float*)d_in[5];
  const float* bk = (const float*)d_in[6];
  const float* Wv = (const float*)d_in[7];
  const float* bv = (const float*)d_in[8];
  const float* Wo = (const float*)d_in[9];
  const float* bo = (const float*)d_in[10];

  float* out     = (float*)d_out;                          // [2,2048,1024]
  float* weights = out + (size_t)2 * SEQ * D_MODEL;        // [2,16,2048,2048]

  // Workspace (bf16 elements): Q | K | V^T | context  == 4 x 4Mi elem = 32 MB
  unsigned short* ws = (unsigned short*)d_ws;
  const size_t NE = (size_t)2 * N_HEADS * SEQ * DEPTH;     // 4,194,304
  void* Qb  = (void*)(ws + 0 * NE);
  void* Kb  = (void*)(ws + 1 * NE);
  void* VTb = (void*)(ws + 2 * NE);
  void* Ab  = (void*)(ws + 3 * NE);

  const int gemm_blocks = (MTOT / 16) * (D_MODEL / 16) / 8; // 2048
  mha_proj_kernel<<<gemm_blocks, 256, 0, stream>>>(q, Wq, bq, Qb, 0);
  mha_proj_kernel<<<gemm_blocks, 256, 0, stream>>>(k, Wk, bk, Kb, 1);
  mha_proj_kernel<<<gemm_blocks, 256, 0, stream>>>(v, Wv, bv, VTb, 2);

  const int attn_blocks = 2 * N_HEADS * (SEQ / 16);         // 4096
  const size_t smem = (size_t)(16 * SEQ + 16 * DEPTH + 256 + 16 + 16) * sizeof(float); // 136320 B
  mha_attn_kernel<<<attn_blocks, 256, smem, stream>>>(Qb, Kb, VTb, weights, Ab);

  mha_out_kernel<<<gemm_blocks, 256, 0, stream>>>(Ab, Wo, bo, out);
}
